// NonLocalModule_79010218377228
// MI455X (gfx1250) — matrix operand order repaired and checked by reading, not compiled, not run on hardware
//
#include <hip/hip_runtime.h>
#include <hip/hip_bf16.h>

typedef __attribute__((ext_vector_type(16))) _Float16 v16h;
typedef __attribute__((ext_vector_type(8)))  float    v8f;

#define BATCH 2
#define CH    3
#define IMH   63
#define IMW   63
#define KW    7
#define NHW   57            // 63-7+1
#define NTOT  (NHW*NHW)     // 3249
#define NPAD  3264          // multiple of 32
#define NT    204           // NPAD/16 (16-row tiles)
#define MCNT  102           // NPAD/32 (32-key chunks)
#define KS    147           // 3*7*7 patch features
#define KSP   160           // padded to multiple of 32
#define KC    5             // KSP/32 reduction chunks
#define KFT   10            // KSP/16 output-feature tiles

// ---------------------------------------------------------------- helpers ----
__device__ __forceinline__ float sample_patch(const float* img, int b, int n, int k) {
    if (n >= NTOT || k >= KS) return 0.f;
    int nh = n / NHW, nw = n % NHW;
    int c  = k / 49, r = k % 49;
    int ky = r / 7,  kx = r % 7;
    return img[((b*CH + c)*IMH + nh + ky)*IMW + nw + kx];
}

__device__ __forceinline__ float gray_at(const float* img, int b, int hh, int ww) {
    if (hh < 0 || hh >= IMH || ww < 0 || ww >= IMW) return 0.f;  // SAME zero pad
    const float* p = img + (size_t)b*CH*IMH*IMW + hh*IMW + ww;
    return 0.299f*p[0] + 0.587f*p[IMH*IMW] + 0.114f*p[2*IMH*IMW];
}

// ------------------------------------------------------- 1) block detect ----
__global__ void block_kernel(const float* __restrict__ x, const float* __restrict__ y,
                             float* __restrict__ blockimg) {
    int t = blockIdx.x*blockDim.x + threadIdx.x;
    if (t >= BATCH*IMH*IMW) return;
    int b = t / (IMH*IMW), r = t % (IMH*IMW);
    int hh = r / IMW, ww = r % IMW;
    float px[9], py[9];
    #pragma unroll
    for (int dy = 0; dy < 3; ++dy)
        #pragma unroll
        for (int dx = 0; dx < 3; ++dx) {
            px[dy*3+dx] = gray_at(x, b, hh-1+dy, ww-1+dx);
            py[dy*3+dx] = gray_at(y, b, hh-1+dy, ww-1+dx);
        }
    float in_gx = -px[0]+px[2] - 2.f*px[3]+2.f*px[5] - px[6]+px[8];
    float in_gy = -px[0]-2.f*px[1]-px[2] + px[6]+2.f*px[7]+px[8];
    float rf_gx = -py[0]+py[2] - 2.f*py[3]+2.f*py[5] - py[6]+py[8];
    float rf_gy = -py[0]-2.f*py[1]-py[2] + py[6]+2.f*py[7]+py[8];
    if (ww == IMW-1) { in_gx = 1.f; rf_gx = 1.f; }   // .at[:,:,:,-1].set(1)
    if (hh == IMH-1) { in_gy = 1.f; rf_gy = 1.f; }   // .at[:,:,-1,:].set(1)
    float n_in = sqrtf(in_gx*in_gx + in_gy*in_gy);
    float n_rf = sqrtf(rf_gx*rf_gx + rf_gy*rf_gy);
    float cosv = (in_gx*rf_gx + in_gy*rf_gy + 0.001f) / (n_in*n_rf + 0.001f);
    float sigma = (n_in >= 29.f) ? 1.f : 0.f;
    float v = sigma * ((n_rf >= 29.f) ? (1.f - cosv) : 1.f);
    blockimg[t] = 1.f - v;
}

// ----------------------------------- 2) per-position sq, 1/(h^2+1), blk_p ----
__global__ void stats_kernel(const float* __restrict__ y,
                             const float* __restrict__ w1, const float* __restrict__ b1,
                             const float* __restrict__ w2, const float* __restrict__ b2,
                             const float* __restrict__ blockimg,
                             float* __restrict__ sq, float* __restrict__ hinv,
                             float* __restrict__ blkp) {
    int t = blockIdx.x*blockDim.x + threadIdx.x;
    if (t >= BATCH*NPAD) return;
    int b = t / NPAD, n = t % NPAD;
    if (n >= NTOT) { sq[t] = 1e30f; hinv[t] = 0.f; blkp[t] = 0.f; return; }
    int nh = n / NHW, nw = n % NHW;
    float ss = 0.f, h0 = b1[0], h1 = b1[1], h2 = b1[2];
    for (int c = 0; c < CH; ++c)
        for (int ky = 0; ky < KW; ++ky)
            for (int kx = 0; kx < KW; ++kx) {
                float v = y[((b*CH + c)*IMH + nh + ky)*IMW + nw + kx];
                int idx = c*49 + ky*7 + kx;
                ss += v*v;
                h0 += w1[idx]       * v;
                h1 += w1[KS + idx]  * v;
                h2 += w1[2*KS+idx]  * v;
            }
    h0 = fmaxf(h0, 0.f); h1 = fmaxf(h1, 0.f); h2 = fmaxf(h2, 0.f);
    float h = b2[0] + w2[0]*h0 + w2[1]*h1 + w2[2]*h2;
    sq[t]   = ss;
    hinv[t] = 1.f / (h*h + 1.f);
    float s = 0.f;
    for (int ky = 0; ky < KW; ++ky)
        for (int kx = 0; kx < KW; ++kx)
            s += blockimg[(b*IMH + nh + ky)*IMW + nw + kx];
    blkp[t] = s * (1.f/49.f);
}

// -------------------- 3) im2col directly into WMMA register layouts (f16) ----
// refA: A-layout (queries)   [B][NT][KC][32 lanes][16 halves]
// refB: B-layout (keys)      [B][NT][KC][32 lanes][16 halves]
// inpB: B-layout for GEMM2   [B][MCNT][KFT][32 lanes][16 halves]
__global__ void fill_layout_kernel(const float* __restrict__ x, const float* __restrict__ y,
                                   _Float16* __restrict__ refA, _Float16* __restrict__ refB,
                                   _Float16* __restrict__ inpB) {
    const int R = BATCH*NT*KC*512;
    const int Q = BATCH*MCNT*KFT*512;
    int tid = blockIdx.x*blockDim.x + threadIdx.x;
    if (tid < R) {
        int hh = tid & 15, lane = (tid >> 4) & 31, rest = tid >> 9;
        int c = rest % KC; rest /= KC;
        int qt = rest % NT; int b = rest / NT;
        int n = qt*16 + (lane & 15);
        int g = lane >> 4;
        int k = c*32 + hh + 8*(g + (hh >> 3));          // ISA A 16x32 f16 layout
        refA[tid] = (_Float16)sample_patch(y, b, n, k);
    } else if (tid < 2*R) {
        int id = tid - R;
        int hh = id & 15, lane = (id >> 4) & 31, rest = id >> 9;
        int c = rest % KC; rest /= KC;
        int mt = rest % NT; int b = rest / NT;
        int m = mt*16 + (lane & 15);
        int k = c*32 + hh + 8*((lane >> 4) + (hh >> 3)); // ISA B 32x16 f16 layout
        refB[id] = (_Float16)sample_patch(y, b, m, k);
    } else if (tid < 2*R + Q) {
        int id = tid - 2*R;
        int hh = id & 15, lane = (id >> 4) & 31, rest = id >> 9;
        int f = rest % KFT; rest /= KFT;
        int mc = rest % MCNT; int b = rest / MCNT;
        int kf = f*16 + (lane & 15);
        int m  = mc*32 + hh + 8*((lane >> 4) + (hh >> 3));
        inpB[id] = (_Float16)sample_patch(x, b, m, kf);
    }
}

// ------------------------- 4) flash non-local attention, dual-GEMM in WMMA ----
__global__ __launch_bounds__(128) void flash_kernel(
        const _Float16* __restrict__ refA, const _Float16* __restrict__ refB,
        const _Float16* __restrict__ inpB,
        const float* __restrict__ sq, const float* __restrict__ hinv,
        const float* __restrict__ blk, float* __restrict__ outp) {
    __shared__ float lds[4][16*32];
    int lane  = threadIdx.x & 31;
    int wv    = threadIdx.x >> 5;
    int wid   = blockIdx.x*4 + wv;
    int b     = wid / NT;
    int qt    = wid % NT;
    float* P  = lds[wv];
    int lm = lane & 15;
    int g  = lane >> 4;

    const v16h* refAv = (const v16h*)refA;
    const v16h* refBv = (const v16h*)refB;
    const v16h* inpBv = (const v16h*)inpB;

    // persistent query tile (A-layout), 5 K-chunks
    v16h qA[KC];
    #pragma unroll
    for (int c = 0; c < KC; ++c) qA[c] = refAv[((b*NT + qt)*KC + c)*32 + lane];

    // per-lane row params: VGPR r <-> row (r + 8*g) of the tile (C/D layout)
    float sqn[8], hr[8], mrow[8], Zr[8], Wr[8];
    #pragma unroll
    for (int r = 0; r < 8; ++r) {
        int n = qt*16 + r + 8*g;
        sqn[r]  = sq[b*NPAD + n];
        hr[r]   = hinv[b*NPAD + n];
        mrow[r] = -1e30f; Zr[r] = 0.f; Wr[r] = 0.f;
    }
    const v8f zero8 = {0.f,0.f,0.f,0.f,0.f,0.f,0.f,0.f};
    v8f acc[KFT];
    #pragma unroll
    for (int f = 0; f < KFT; ++f) acc[f] = zero8;

    for (int mc = 0; mc < MCNT; ++mc) {
        int mt0 = mc*2, mt1 = mc*2 + 1;
        v8f S0 = zero8, S1 = zero8;
        #pragma unroll
        for (int c = 0; c < KC; ++c) {
            v16h kb = refBv[((b*NT + mt0)*KC + c)*32 + lane];
            S0 = __builtin_amdgcn_wmma_f32_16x16x32_f16(false, qA[c], false, kb,
                                                        (short)0, S0, false, false);
        }
        #pragma unroll
        for (int c = 0; c < KC; ++c) {
            v16h kb = refBv[((b*NT + mt1)*KC + c)*32 + lane];
            S1 = __builtin_amdgcn_wmma_f32_16x16x32_f16(false, qA[c], false, kb,
                                                        (short)0, S1, false, false);
        }
        float sqm0 = sq [b*NPAD + mt0*16 + lm];
        float sqm1 = sq [b*NPAD + mt1*16 + lm];
        float bk0  = blk[b*NPAD + mt0*16 + lm];
        float bk1  = blk[b*NPAD + mt1*16 + lm];

        float s0[8], s1[8], nm[8];
        #pragma unroll
        for (int r = 0; r < 8; ++r) {
            s0[r] = (2.f*S0[r] - sqn[r] - sqm0) * hr[r];
            s1[r] = (2.f*S1[r] - sqn[r] - sqm1) * hr[r];
            float mx = fmaxf(s0[r], s1[r]);
            #pragma unroll
            for (int off = 1; off < 16; off <<= 1)
                mx = fmaxf(mx, __shfl_xor(mx, off, 32));   // stays inside 16-lane group
            nm[r] = fmaxf(mrow[r], mx);
        }
        #pragma unroll
        for (int r = 0; r < 8; ++r) {
            float scale = __expf(mrow[r] - nm[r]);
            mrow[r] = nm[r];
            float p0 = __expf(s0[r] - nm[r]);
            float p1 = __expf(s1[r] - nm[r]);
            float pz = p0 + p1;
            float pw = p0*bk0 + p1*bk1;
            #pragma unroll
            for (int off = 1; off < 16; off <<= 1) {
                pz += __shfl_xor(pz, off, 32);
                pw += __shfl_xor(pw, off, 32);
            }
            Zr[r] = Zr[r]*scale + pz;
            Wr[r] = Wr[r]*scale + pw;
            #pragma unroll
            for (int f = 0; f < KFT; ++f) acc[f][r] *= scale;
            P[(r + 8*g)*32 + lm]      = p0*bk0;            // masked probs, pre-norm
            P[(r + 8*g)*32 + 16 + lm] = p1*bk1;
        }
        __syncthreads();
        // transpose P (C/D layout) -> A-layout f16 tile: row lm, 32 keys
        v16h aP;
        #pragma unroll
        for (int j = 0; j < 8; ++j) {
            int kl = 2*j + 8*(g + ((j >= 4) ? 1 : 0));
            aP[2*j]   = (_Float16)P[lm*32 + kl];
            aP[2*j+1] = (_Float16)P[lm*32 + kl + 1];
        }
        __syncthreads();
        #pragma unroll
        for (int f = 0; f < KFT; ++f) {
            v16h bI = inpBv[((b*MCNT + mc)*KFT + f)*32 + lane];
            acc[f] = __builtin_amdgcn_wmma_f32_16x16x32_f16(false, aP, false, bI,
                                                            (short)0, acc[f], false, false);
        }
    }
    // att = e*blk / (W + 0.001*Z)  (same running-max normalization throughout)
    #pragma unroll
    for (int r = 0; r < 8; ++r) {
        int n = qt*16 + r + 8*g;
        if (n >= NTOT) continue;
        float inv = 1.f / (Wr[r] + 0.001f*Zr[r]);
        #pragma unroll
        for (int f = 0; f < KFT; ++f) {
            int kf = f*16 + lm;
            outp[((size_t)b*KSP + kf)*NPAD + n] = acc[f][r] * inv;
        }
    }
}

// ------------------------------------------- 5) overlap-add fold + weights ----
__global__ void fold_kernel(const float* __restrict__ outp, float* __restrict__ out) {
    int t = blockIdx.x*blockDim.x + threadIdx.x;
    if (t >= BATCH*CH*IMH*IMW) return;
    int b = t / (CH*IMH*IMW), r = t % (CH*IMH*IMW);
    int c = r / (IMH*IMW); r %= IMH*IMW;
    int hh = r / IMW, ww = r % IMW;
    float acc = 0.f, cnt = 0.f;
    for (int ky = 0; ky < KW; ++ky)
        for (int kx = 0; kx < KW; ++kx) {
            int nh = hh - ky, nw = ww - kx;
            if (nh >= 0 && nh < NHW && nw >= 0 && nw < NHW) {
                int cf = c*49 + ky*7 + kx;
                acc += outp[((size_t)b*KSP + cf)*NPAD + nh*NHW + nw];
                cnt += 1.f;
            }
        }
    out[t] = acc / cnt;
}

// ----------------------------------------------------------------- launch ----
extern "C" void kernel_launch(void* const* d_in, const int* in_sizes, int n_in,
                              void* d_out, int out_size, void* d_ws, size_t ws_size,
                              hipStream_t stream) {
    const float* x  = (const float*)d_in[0];
    const float* y  = (const float*)d_in[1];
    const float* w1 = (const float*)d_in[2];
    const float* b1 = (const float*)d_in[3];
    const float* w2 = (const float*)d_in[4];
    const float* b2 = (const float*)d_in[5];
    float* out = (float*)d_out;

    char* ws = (char*)d_ws;
    size_t off = 0;
    auto carve = [&](size_t bytes) -> char* {
        char* p = ws + off;
        off = (off + bytes + 255) & ~(size_t)255;
        return p;
    };
    const size_t R = (size_t)BATCH*NT*KC*512;     // halves in refA / refB
    const size_t Q = (size_t)BATCH*MCNT*KFT*512;  // halves in inpB
    _Float16* refA = (_Float16*)carve(R*2);
    _Float16* refB = (_Float16*)carve(R*2);
    _Float16* inpB = (_Float16*)carve(Q*2);
    float* sqb   = (float*)carve((size_t)BATCH*NPAD*4);
    float* hinvb = (float*)carve((size_t)BATCH*NPAD*4);
    float* blkb  = (float*)carve((size_t)BATCH*NPAD*4);
    float* bimg  = (float*)carve((size_t)BATCH*IMH*IMW*4);
    float* outp  = (float*)carve((size_t)BATCH*KSP*NPAD*4);

    {   // 1) block-detect image
        int n = BATCH*IMH*IMW;
        block_kernel<<<(n+255)/256, 256, 0, stream>>>(x, y, bimg);
    }
    {   // 2) sq / bandwidth / block-patch mean
        int n = BATCH*NPAD;
        stats_kernel<<<(n+255)/256, 256, 0, stream>>>(y, w1, b1, w2, b2, bimg,
                                                      sqb, hinvb, blkb);
    }
    {   // 3) WMMA-layout im2col fills
        size_t n = 2*R + Q;
        fill_layout_kernel<<<(int)((n+255)/256), 256, 0, stream>>>(x, y, refA, refB, inpB);
    }
    {   // 4) flash attention: (B*NT) waves, 4 waves per block
        int blocks = (BATCH*NT)/4;    // 408/4 = 102
        flash_kernel<<<blocks, 128, 0, stream>>>(refA, refB, inpB,
                                                 sqb, hinvb, blkb, outp);
    }
    {   // 5) fold to output image
        int n = BATCH*CH*IMH*IMW;
        fold_kernel<<<(n+255)/256, 256, 0, stream>>>(outp, out);
    }
}
